// Qwen3Block_69320772158425
// MI455X (gfx1250) — matrix operand order repaired and checked by reading, not compiled, not run on hardware
//
#include <hip/hip_runtime.h>

#define B_  2
#define S_  2048
#define E_  2048
#define H_  16
#define KV_ 8
#define HD_ 128
#define HID_ 6144
#define EPS_ 1e-6f

typedef __bf16 bf16_t;
typedef __attribute__((ext_vector_type(8)))  __bf16 bf16x8;
typedef __attribute__((ext_vector_type(16))) __bf16 bf16x16;
typedef __attribute__((ext_vector_type(8)))  float  floatx8;
typedef __attribute__((ext_vector_type(4)))  unsigned int u32x4;
typedef __attribute__((ext_vector_type(8)))  int          i32x8;
typedef __attribute__((ext_vector_type(4)))  int          i32x4;

#if defined(__HIP_DEVICE_COMPILE__) && defined(__gfx1250__) && \
    __has_builtin(__builtin_amdgcn_tensor_load_to_lds) &&      \
    __has_builtin(__builtin_amdgcn_s_wait_tensorcnt)
#define USE_TDM 1
#else
#define USE_TDM 0
#endif

__device__ __forceinline__ floatx8 fzero8() {
    floatx8 r;
#pragma unroll
    for (int i = 0; i < 8; ++i) r[i] = 0.f;
    return r;
}

__device__ __forceinline__ bf16x16 cat8(bf16x8 lo, bf16x8 hi) {
    bf16x16 r;
#pragma unroll
    for (int i = 0; i < 8; ++i) { r[i] = lo[i]; r[i + 8] = hi[i]; }
    return r;
}

// A-operand fragment (16x32 bf16). p = &A[row_of_lane][k0] (lane row = lane&15).
// ISA layout: lanes 0-15 hold K 0..7 / 16..23, lanes 16-31 hold K 8..15 / 24..31.
__device__ __forceinline__ bf16x16 frag_a(const bf16_t* p) {
    const int half = (threadIdx.x & 31) >> 4;
    bf16x8 lo = *(const bf16x8*)(p + half * 8);
    bf16x8 hi = *(const bf16x8*)(p + 16 + half * 8);
    return cat8(lo, hi);
}

// B-operand fragment (32x16 bf16) from K-major storage.
// p = &Bt[col_of_lane][k0]; lanes 0-15 K 0..7 / 16..23, lanes 16-31 K 8..15 / 24..31.
__device__ __forceinline__ bf16x16 frag_b(const bf16_t* p) {
    const int half = (threadIdx.x & 31) >> 4;
    bf16x8 lo = *(const bf16x8*)(p + half * 8);
    bf16x8 hi = *(const bf16x8*)(p + 16 + half * 8);
    return cat8(lo, hi);
}

__device__ __forceinline__ floatx8 wmma_bf16(bf16x16 a, bf16x16 b, floatx8 c) {
    return __builtin_amdgcn_wmma_f32_16x16x32_bf16(false, a, false, b, (short)0, c,
                                                   false, false);
}

#if USE_TDM
// Issue a TDM load of a 2D bf16 tile (tile_d0 contiguous elems x tile_d1 rows,
// row stride = stride0 elems) into contiguous LDS at lds_off.
// D# bitfields per CDNA5 ISA ch.8 (group0: count/lds/global/type, group1: dims).
__device__ __forceinline__ void tdm_load_2d(const void* gaddr, unsigned lds_off,
                                            unsigned tile_d0, unsigned tile_d1,
                                            unsigned tensor_d0, unsigned tensor_d1,
                                            unsigned long long stride0) {
    const unsigned long long ga = (unsigned long long)gaddr;
    u32x4 g0;
    g0[0] = 1u;                                     // count=1, user mode
    g0[1] = lds_off;                                // lds_addr (bytes)
    g0[2] = (unsigned)(ga & 0xffffffffu);           // global_addr[31:0]
    g0[3] = (unsigned)((ga >> 32) & 0x1ffffffu) |   // global_addr[56:32]
            (2u << 30);                             // type=2 ("image")
    i32x8 g1;
    g1[0] = 0x10000;                                // data_size=1 -> 2 bytes
    g1[1] = (int)((tensor_d0 & 0xffffu) << 16);     // tensor_dim0[15:0]
    g1[2] = (int)((tensor_d0 >> 16) | ((tensor_d1 & 0xffffu) << 16));
    g1[3] = (int)((tensor_d1 >> 16) | (tile_d0 << 16));  // tile_dim0
    g1[4] = (int)(tile_d1 & 0xffffu);               // tile_dim1 (tile_dim2=0)
    g1[5] = (int)(stride0 & 0xffffffffull);         // tensor_dim0_stride[31:0]
    g1[6] = (int)((stride0 >> 32) & 0xffffull);     // tensor_dim0_stride[47:32]
    g1[7] = 0;
    i32x4 g2 = {0, 0, 0, 0};
    i32x4 g3 = {0, 0, 0, 0};
#if __clang_major__ >= 23
    i32x8 g4 = {0, 0, 0, 0, 0, 0, 0, 0};
    __builtin_amdgcn_tensor_load_to_lds(g0, g1, g2, g3, g4, 0);
#else
    __builtin_amdgcn_tensor_load_to_lds(g0, g1, g2, g3, 0);
#endif
}
#endif

// ---------------------------------------------------------------------------
// Weight prep: fp32 [K][N] -> bf16 transposed [N][K] (one-time; lands in L2).
// ---------------------------------------------------------------------------
__global__ void transpose_cvt_kernel(const float* __restrict__ W, bf16_t* __restrict__ Wt,
                                     int K, int N) {
    __shared__ float t[32][33];
    const int n0 = blockIdx.x * 32, k0 = blockIdx.y * 32;
    const int tx = threadIdx.x, ty = threadIdx.y; // block (32, 8)
#pragma unroll
    for (int j = 0; j < 32; j += 8)
        t[ty + j][tx] = W[(size_t)(k0 + ty + j) * N + n0 + tx];
    __syncthreads();
#pragma unroll
    for (int j = 0; j < 32; j += 8)
        Wt[(size_t)(n0 + ty + j) * K + k0 + tx] = (bf16_t)t[tx][ty + j];
}

// ---------------------------------------------------------------------------
// RMSNorm fp32 row -> bf16 row (E = 2048, block = 256 threads, 8 elems/thread)
// ---------------------------------------------------------------------------
__global__ void rmsnorm_kernel(const float* __restrict__ x, const float* __restrict__ w,
                               bf16_t* __restrict__ out) {
    const int row = blockIdx.x;
    const int tid = threadIdx.x;
    const float* xr = x + (size_t)row * E_;
    float v[8];
    float ss = 0.f;
#pragma unroll
    for (int i = 0; i < 8; ++i) {
        v[i] = xr[tid + i * 256];
        ss += v[i] * v[i];
    }
#pragma unroll
    for (int m = 16; m >= 1; m >>= 1) ss += __shfl_xor(ss, m, 32);
    __shared__ float red[8];
    __shared__ float s_r;
    if ((tid & 31) == 0) red[tid >> 5] = ss;
    __syncthreads();
    if (tid == 0) {
        float t = 0.f;
#pragma unroll
        for (int i = 0; i < 8; ++i) t += red[i];
        s_r = rsqrtf(t / (float)E_ + EPS_);
    }
    __syncthreads();
    const float r = s_r;
#pragma unroll
    for (int i = 0; i < 8; ++i) {
        const int c = tid + i * 256;
        out[(size_t)row * E_ + c] = (bf16_t)(v[i] * r * w[c]);
    }
}

// ---------------------------------------------------------------------------
// Generic bf16 WMMA GEMM: C[M,N] = A[M,K] * Bt[N,K]^T (+ residual).
// 128x128 tile, 8 waves (4x2), each wave 32x64 via 2x4 WMMA accumulators,
// BK=32, double-buffered LDS. When available, tiles are streamed by the
// Tensor Data Mover (wave 0 issues descriptors; TENSORcnt fences the tile).
// ---------------------------------------------------------------------------
template <bool ADD_RES, bool OUT_BF16>
__global__ __launch_bounds__(256, 2) void gemm_bf16_kernel(
    const bf16_t* __restrict__ A, const bf16_t* __restrict__ Bt, void* __restrict__ C,
    const float* __restrict__ Res, int M, int N, int K) {
    __shared__ bf16_t As[2][128 * 32];
    __shared__ bf16_t Bs[2][128 * 32];
    const int bm = blockIdx.x * 128, bn = blockIdx.y * 128;
    const int tid = threadIdx.x, wid = tid >> 5, lane = tid & 31;
    const int wr = wid >> 1, wc = wid & 1;
    const int half = lane >> 4, col = lane & 15, rowl = lane & 15;
    const int nk = K >> 5;

    floatx8 acc[2][4];
#pragma unroll
    for (int mi = 0; mi < 2; ++mi)
#pragma unroll
        for (int ni = 0; ni < 4; ++ni) acc[mi][ni] = fzero8();

#if USE_TDM
    const unsigned asOff0 = (unsigned)(uintptr_t)&As[0][0];
    const unsigned asOff1 = (unsigned)(uintptr_t)&As[1][0];
    const unsigned bsOff0 = (unsigned)(uintptr_t)&Bs[0][0];
    const unsigned bsOff1 = (unsigned)(uintptr_t)&Bs[1][0];
    auto issue_tile = [&](int kt, int buf) {
        tdm_load_2d(A + (size_t)bm * K + kt * 32, buf ? asOff1 : asOff0,
                    32, 128, (unsigned)K, (unsigned)M, (unsigned long long)K);
        tdm_load_2d(Bt + (size_t)bn * K + kt * 32, buf ? bsOff1 : bsOff0,
                    32, 128, (unsigned)K, (unsigned)N, (unsigned long long)K);
    };
    if (wid == 0) {
        issue_tile(0, 0);
        __builtin_amdgcn_s_wait_tensorcnt(0);
    }
#else
    auto load_tile = [&](int kt, int buf) {
#pragma unroll
        for (int i = tid; i < 512; i += 256) {
            const int r = i >> 2, c = (i & 3) * 8;
            *(bf16x8*)&As[buf][r * 32 + c] =
                *(const bf16x8*)&A[(size_t)(bm + r) * K + kt * 32 + c];
            *(bf16x8*)&Bs[buf][r * 32 + c] =
                *(const bf16x8*)&Bt[(size_t)(bn + r) * K + kt * 32 + c];
        }
    };
    load_tile(0, 0);
#endif
    __syncthreads();

    int buf = 0;
    for (int kt = 0; kt < nk; ++kt) {
#if USE_TDM
        if (kt + 1 < nk && wid == 0) issue_tile(kt + 1, buf ^ 1);
#else
        if (kt + 1 < nk) load_tile(kt + 1, buf ^ 1);
#endif
        bf16x16 aF[2], bF[4];
#pragma unroll
        for (int mi = 0; mi < 2; ++mi)
            aF[mi] = frag_a(&As[buf][(wr * 32 + mi * 16 + rowl) * 32]);
#pragma unroll
        for (int ni = 0; ni < 4; ++ni)
            bF[ni] = frag_b(&Bs[buf][(wc * 64 + ni * 16 + col) * 32]);
#pragma unroll
        for (int mi = 0; mi < 2; ++mi)
#pragma unroll
            for (int ni = 0; ni < 4; ++ni)
                acc[mi][ni] = wmma_bf16(aF[mi], bF[ni], acc[mi][ni]);
#if USE_TDM
        if (wid == 0) __builtin_amdgcn_s_wait_tensorcnt(0);
#endif
        __syncthreads();
        buf ^= 1;
    }

#pragma unroll
    for (int mi = 0; mi < 2; ++mi)
#pragma unroll
        for (int ni = 0; ni < 4; ++ni)
#pragma unroll
            for (int r = 0; r < 8; ++r) {
                const int gm = bm + wr * 32 + mi * 16 + 8 * half + r;
                const int gn = bn + wc * 64 + ni * 16 + col;
                float v = acc[mi][ni][r];
                if (ADD_RES) v += Res[(size_t)gm * N + gn];
                if (OUT_BF16)
                    ((bf16_t*)C)[(size_t)gm * N + gn] = (bf16_t)v;
                else
                    ((float*)C)[(size_t)gm * N + gn] = v;
            }
}

// ---------------------------------------------------------------------------
// QKV postprocess: per (b,s,head32) 128-vector: rmsnorm (q/k) + RoPE (q/k),
// plain bf16 convert (v). Block = 128 threads.
// ---------------------------------------------------------------------------
__global__ void qkv_prep_kernel(const float* __restrict__ qkv, const float* __restrict__ qn,
                                const float* __restrict__ kn, bf16_t* __restrict__ Qo,
                                bf16_t* __restrict__ Ko, bf16_t* __restrict__ Vo) {
    const int blk = blockIdx.x;
    const int bs = blk >> 5;       // 0..B*S-1
    const int hh = blk & 31;       // 0..31 (16 q + 8 k + 8 v)
    const int s = bs & (S_ - 1);
    const int tid = threadIdx.x;
    const float* rowp = qkv + (size_t)bs * ((H_ + 2 * KV_) * HD_) + hh * HD_;
    const float v = rowp[tid];

    float ss = v * v;
#pragma unroll
    for (int m = 16; m >= 1; m >>= 1) ss += __shfl_xor(ss, m, 32);
    __shared__ float red[4];
    __shared__ float srms;
    __shared__ float buf[128];
    if ((tid & 31) == 0) red[tid >> 5] = ss;
    __syncthreads();
    if (tid == 0) srms = rsqrtf((red[0] + red[1] + red[2] + red[3]) / (float)HD_ + EPS_);
    __syncthreads();

    bf16_t* outp;
    if (hh < H_)
        outp = Qo + ((size_t)bs * H_ + hh) * HD_;
    else if (hh < H_ + KV_)
        outp = Ko + ((size_t)bs * KV_ + (hh - H_)) * HD_;
    else
        outp = Vo + ((size_t)bs * KV_ + (hh - H_ - KV_)) * HD_;

    if (hh >= H_ + KV_) {   // V: passthrough convert (block-uniform branch)
        outp[tid] = (bf16_t)v;
        return;
    }
    const float nw = (hh < H_) ? qn[tid] : kn[tid];
    buf[tid] = v * srms * nw;
    __syncthreads();
    if (tid < 64) {
        const float x1 = buf[2 * tid], x2 = buf[2 * tid + 1];
        const float inv = __powf(1000000.0f, -((float)(2 * tid)) / (float)HD_);
        const float ang = (float)s * inv;
        float sn, cs;
        __sincosf(ang, &sn, &cs);
        outp[2 * tid]     = (bf16_t)(x1 * cs - x2 * sn);
        outp[2 * tid + 1] = (bf16_t)(x1 * sn + x2 * cs);
    }
}

// ---------------------------------------------------------------------------
// Flash attention (causal, GQA G=2). Grid (S/128, H, B), block 256 (8 waves).
// Each wave owns 16 q rows; Q kept in registers as 4 A-fragments; K/V tiles of
// 32 kv positions staged in LDS; P relaid through LDS into A-operand layout.
// ---------------------------------------------------------------------------
__global__ __launch_bounds__(256, 1) void flash_attn_kernel(
    const bf16_t* __restrict__ Q, const bf16_t* __restrict__ K,
    const bf16_t* __restrict__ V, bf16_t* __restrict__ O) {
    const int qb = blockIdx.x, h = blockIdx.y, b = blockIdx.z;
    const int kvh = h >> 1; // G = 2
    const int tid = threadIdx.x, wid = tid >> 5, lane = tid & 31;
    const int half = lane >> 4, col = lane & 15, rowl = lane & 15;

    __shared__ bf16_t Ks[32 * 128]; // [c][d] : B-operand for Q*K^T
    __shared__ bf16_t Vs[128 * 32]; // [d][c] : B-operand for P*V (transposed)
    __shared__ bf16_t Ps[128 * 32]; // per-wave P staging, A layout source

    bf16x16 qF[4];
    {
        const bf16_t* qrow =
            Q + (((size_t)b * S_ + qb * 128 + wid * 16 + rowl) * H_ + h) * HD_;
#pragma unroll
        for (int kc = 0; kc < 4; ++kc) qF[kc] = frag_a(qrow + kc * 32);
    }

    floatx8 oacc[8];
#pragma unroll
    for (int i = 0; i < 8; ++i) oacc[i] = fzero8();
    float mr[8], lr[8];
#pragma unroll
    for (int r = 0; r < 8; ++r) { mr[r] = -1e30f; lr[r] = 0.f; }

    const float scale = 0.08838834764831845f; // 1/sqrt(128)
    const int steps = (qb + 1) * 4;

    for (int st = 0; st < steps; ++st) {
        const int c0 = st * 32;
        __syncthreads();
        // K tile (coalesced, row-major [c][d])
        for (int i = tid; i < 512; i += 256) {
            const int cc = i >> 4, d = (i & 15) * 8;
            *(bf16x8*)&Ks[cc * 128 + d] =
                *(const bf16x8*)&K[(((size_t)b * S_ + c0 + cc) * KV_ + kvh) * HD_ + d];
        }
        // V tile, transposed into [d][c]
        for (int i = tid; i < 512; i += 256) {
            const int cc = i >> 4, d0 = (i & 15) * 8;
            bf16x8 t =
                *(const bf16x8*)&V[(((size_t)b * S_ + c0 + cc) * KV_ + kvh) * HD_ + d0];
#pragma unroll
            for (int j = 0; j < 8; ++j) Vs[(d0 + j) * 32 + cc] = t[j];
        }
        __syncthreads();

        // S = Q * K^T : 2 N-tiles x 4 chained K-steps
        floatx8 sc[2];
        sc[0] = fzero8();
        sc[1] = fzero8();
#pragma unroll
        for (int nt = 0; nt < 2; ++nt) {
            const bf16_t* bbase = &Ks[(nt * 16 + col) * 128];
#pragma unroll
            for (int kc = 0; kc < 4; ++kc)
                sc[nt] = wmma_bf16(qF[kc], frag_b(bbase + kc * 32), sc[nt]);
        }

        // online softmax
        float mnew[8];
#pragma unroll
        for (int r = 0; r < 8; ++r) {
            const int qr = qb * 128 + wid * 16 + 8 * half + r;
            float s0 = sc[0][r] * scale;
            float s1 = sc[1][r] * scale;
            if (c0 + col > qr) s0 = -1e30f;
            if (c0 + 16 + col > qr) s1 = -1e30f;
            sc[0][r] = s0;
            sc[1][r] = s1;
            float t = fmaxf(s0, s1);
            t = fmaxf(t, __shfl_xor(t, 1, 32));
            t = fmaxf(t, __shfl_xor(t, 2, 32));
            t = fmaxf(t, __shfl_xor(t, 4, 32));
            t = fmaxf(t, __shfl_xor(t, 8, 32));
            mnew[r] = fmaxf(mr[r], t);
        }
#pragma unroll
        for (int r = 0; r < 8; ++r) {
            const float f = __expf(mr[r] - mnew[r]);
            const float p0 = __expf(sc[0][r] - mnew[r]);
            const float p1 = __expf(sc[1][r] - mnew[r]);
            float rs = p0 + p1;
            rs += __shfl_xor(rs, 1, 32);
            rs += __shfl_xor(rs, 2, 32);
            rs += __shfl_xor(rs, 4, 32);
            rs += __shfl_xor(rs, 8, 32);
            lr[r] = lr[r] * f + rs;
            mr[r] = mnew[r];
#pragma unroll
            for (int nt = 0; nt < 8; ++nt) oacc[nt][r] *= f;
            const int prow = wid * 16 + 8 * half + r;
            Ps[prow * 32 + col]      = (bf16_t)p0;
            Ps[prow * 32 + 16 + col] = (bf16_t)p1;
        }
        // own-wave LDS region: in-order per wave, no barrier needed
        bf16x16 pF = frag_a(&Ps[(wid * 16 + rowl) * 32]);
#pragma unroll
        for (int nt = 0; nt < 8; ++nt)
            oacc[nt] = wmma_bf16(pF, frag_b(&Vs[(nt * 16 + col) * 32]), oacc[nt]);
    }

#pragma unroll
    for (int nt = 0; nt < 8; ++nt)
#pragma unroll
        for (int r = 0; r < 8; ++r) {
            const int qr = qb * 128 + wid * 16 + 8 * half + r;
            const int d = nt * 16 + col;
            O[(((size_t)b * S_ + qr) * H_ + h) * HD_ + d] = (bf16_t)(oacc[nt][r] / lr[r]);
        }
}

// ---------------------------------------------------------------------------
// SwiGLU: act = silu(g[:, :HID]) * g[:, HID:]
// ---------------------------------------------------------------------------
__global__ void silu_mul_kernel(const bf16_t* __restrict__ g, bf16_t* __restrict__ act,
                                long total) {
    for (long idx = (long)blockIdx.x * 256 + threadIdx.x; idx < total;
         idx += (long)gridDim.x * 256) {
        const long m = idx / HID_, j = idx - m * HID_;
        const float gv = (float)g[m * (2 * HID_) + j];
        const float uv = (float)g[m * (2 * HID_) + HID_ + j];
        act[idx] = (bf16_t)((gv / (1.f + __expf(-gv))) * uv);
    }
}

// ---------------------------------------------------------------------------
extern "C" void kernel_launch(void* const* d_in, const int* in_sizes, int n_in,
                              void* d_out, int out_size, void* d_ws, size_t ws_size,
                              hipStream_t stream) {
    (void)in_sizes; (void)n_in; (void)out_size; (void)ws_size;
    const float* x       = (const float*)d_in[0];
    const float* ln_w    = (const float*)d_in[1];
    const float* qkv_w   = (const float*)d_in[2];
    const float* qn_w    = (const float*)d_in[3];
    const float* kn_w    = (const float*)d_in[4];
    const float* dense_w = (const float*)d_in[5];
    const float* ffln_w  = (const float*)d_in[6];
    const float* wg1_w   = (const float*)d_in[7];
    const float* w2_w    = (const float*)d_in[8];
    float* out = (float*)d_out;

    const size_t M = (size_t)B_ * S_; // 4096
    char* w = (char*)d_ws;
    auto alloc = [&](size_t bytes) {
        char* p = w;
        w += (bytes + 255) & ~(size_t)255;
        return p;
    };
    bf16_t* qkvT   = (bf16_t*)alloc((size_t)4096 * E_ * 2);
    bf16_t* denseT = (bf16_t*)alloc((size_t)E_ * 2048 * 2);
    bf16_t* wg1T   = (bf16_t*)alloc((size_t)2 * HID_ * E_ * 2);
    bf16_t* w2T    = (bf16_t*)alloc((size_t)E_ * HID_ * 2);
    bf16_t* h_bf   = (bf16_t*)alloc(M * E_ * 2);
    float*  qkv_f  = (float*)alloc(M * 4096 * 4);
    bf16_t* q_bf   = (bf16_t*)alloc(M * H_ * HD_ * 2);
    bf16_t* k_bf   = (bf16_t*)alloc(M * KV_ * HD_ * 2);
    bf16_t* v_bf   = (bf16_t*)alloc(M * KV_ * HD_ * 2);
    bf16_t* attn_bf= (bf16_t*)alloc(M * E_ * 2);
    float*  x1_f   = (float*)alloc(M * E_ * 4);
    bf16_t* h2_bf  = (bf16_t*)alloc(M * E_ * 2);
    bf16_t* g_bf   = (bf16_t*)alloc(M * 2 * HID_ * 2);
    bf16_t* act_bf = (bf16_t*)alloc(M * HID_ * 2);

    const dim3 tb(32, 8);
    // Weight prep (one-time; bf16 weights then live in the 192MB L2)
    transpose_cvt_kernel<<<dim3(4096 / 32, E_ / 32), tb, 0, stream>>>(qkv_w, qkvT, E_, 4096);
    transpose_cvt_kernel<<<dim3(2048 / 32, E_ / 32), tb, 0, stream>>>(dense_w, denseT, E_, 2048);
    transpose_cvt_kernel<<<dim3((2 * HID_) / 32, E_ / 32), tb, 0, stream>>>(wg1_w, wg1T, E_, 2 * HID_);
    transpose_cvt_kernel<<<dim3(E_ / 32, HID_ / 32), tb, 0, stream>>>(w2_w, w2T, HID_, E_);

    // Attention half
    rmsnorm_kernel<<<M, 256, 0, stream>>>(x, ln_w, h_bf);
    gemm_bf16_kernel<false, false><<<dim3(M / 128, 4096 / 128), 256, 0, stream>>>(
        h_bf, qkvT, qkv_f, nullptr, (int)M, 4096, E_);
    qkv_prep_kernel<<<M * 32, 128, 0, stream>>>(qkv_f, qn_w, kn_w, q_bf, k_bf, v_bf);
    flash_attn_kernel<<<dim3(S_ / 128, H_, B_), 256, 0, stream>>>(q_bf, k_bf, v_bf, attn_bf);
    gemm_bf16_kernel<true, false><<<dim3(M / 128, E_ / 128), 256, 0, stream>>>(
        attn_bf, denseT, x1_f, x, (int)M, E_, 2048);

    // MLP half
    rmsnorm_kernel<<<M, 256, 0, stream>>>(x1_f, ffln_w, h2_bf);
    gemm_bf16_kernel<false, true><<<dim3(M / 128, (2 * HID_) / 128), 256, 0, stream>>>(
        h2_bf, wg1T, g_bf, nullptr, (int)M, 2 * HID_, E_);
    const long act_total = (long)M * HID_;
    silu_mul_kernel<<<(unsigned)((act_total + 255) / 256), 256, 0, stream>>>(g_bf, act_bf, act_total);
    gemm_bf16_kernel<true, false><<<dim3(M / 128, E_ / 128), 256, 0, stream>>>(
        act_bf, w2T, out, x1_f, (int)M, E_, HID_);
}
